// improved_3part_route_noact_real_moe_2353642078589
// MI455X (gfx1250) — matrix operand order repaired and checked by reading, not compiled, not run on hardware
//
#include <hip/hip_runtime.h>
#include <hip/hip_bf16.h>
#include <stdint.h>

// Problem constants (from reference)
#define E_  8
#define K_  2
#define T_  16384
#define TK_ 32768
#define H_  512
#define R0_ 256
#define R1_ 256
#define O_  512

typedef unsigned short u16;
typedef __attribute__((ext_vector_type(16))) __bf16 v16bf;
typedef __attribute__((ext_vector_type(8)))  float  v8f;

union V32B { uint4 u[2]; v16bf v; };

static __device__ __forceinline__ u16 f32_to_bf16(float f) {
  union { float f; uint32_t u; } x; x.f = f;
  uint32_t u = x.u;
  uint32_t r = u + 0x7FFFu + ((u >> 16) & 1u);      // round-to-nearest-even
  if ((u & 0x7F800000u) == 0x7F800000u) r = u;       // keep inf/nan pattern
  return (u16)(r >> 16);
}

// ---------------------------------------------------------------------------
// f32 -> bf16 elementwise conversion (x and weight tensors, done once)
// ---------------------------------------------------------------------------
__global__ void k_cvt_bf16(const float* __restrict__ in, u16* __restrict__ out, int n) {
  int i = blockIdx.x * blockDim.x + threadIdx.x;
  if (i < n) out[i] = f32_to_bf16(in[i]);
}

// ---------------------------------------------------------------------------
// Deterministic per-expert slot-row compaction.
// Slot s = t*K + k is routed to expert e iff mask[e,k,t] (one-hot over e).
// Block e scans all TK_ slots in order with a shared-memory inclusive scan.
// ---------------------------------------------------------------------------
__global__ void k_build_lists(const unsigned char* __restrict__ mask,  // [E,K,T] bool
                              int* __restrict__ idx,                   // [E, TK]
                              int* __restrict__ counts) {              // [E]
  __shared__ int sbuf[256];
  __shared__ int sbase;
  const int e = blockIdx.x;
  const int tid = threadIdx.x;
  if (tid == 0) sbase = 0;
  __syncthreads();
  for (int chunk = 0; chunk < TK_; chunk += 256) {
    int s = chunk + tid;
    int t = s >> 1, k = s & 1;
    int m = (int)mask[((size_t)e * K_ + k) * T_ + t];
    sbuf[tid] = m;
    __syncthreads();
    #pragma unroll
    for (int off = 1; off < 256; off <<= 1) {
      int add = (tid >= off) ? sbuf[tid - off] : 0;
      __syncthreads();
      sbuf[tid] += add;
      __syncthreads();
    }
    int incl  = sbuf[tid];
    int total = sbuf[255];
    int pos   = sbase + incl - m;
    if (m) idx[(size_t)e * TK_ + pos] = s;
    __syncthreads();
    if (tid == 0) sbase += total;
    __syncthreads();
  }
  if (tid == 0) counts[e] = sbase;
}

// ---------------------------------------------------------------------------
// Gathered per-expert GEMM using v_wmma_f32_16x16x32_bf16.
// Register-blocked: each wave computes a 64x64 output block (4x4 grid of
// 16x16 tiles) -> 8 A-tile + 8 B-tile b128 loads feed 16 WMMAs per k-step
// (1.0 b128/WMMA vs 2.5 for a 16x64 wave tile).
// Block = 4 waves covering 256 gathered rows x 64 cols.
// A/B operand striping per CDNA5 ISA 7.12.2.
// ---------------------------------------------------------------------------
__global__ __launch_bounds__(128)
void k_moe_gemm(const u16* __restrict__ Abf,     // [T, Kd] bf16 token-major
                const u16* __restrict__ Wbf,     // [E, N, Kd] bf16
                const float* __restrict__ bias,  // [E, N]
                const int* __restrict__ idx,     // [E, TK]
                const int* __restrict__ counts,  // [E]
                float* __restrict__ slotY,       // [TK, N]
                int Kd, int N, int do_relu) {
  const int e    = blockIdx.z;
  const int cnt  = counts[e];
  const int wave = threadIdx.x >> 5;
  const int lane = threadIdx.x & 31;
  const int rowBase = (blockIdx.x * 4 + wave) * 64;   // 64 rows per wave
  if (rowBase >= cnt) return;
  const int nBase = blockIdx.y * 64;
  const int half  = lane >> 4;       // which half-wave
  const int l16   = lane & 15;
  const int* elist = idx + (size_t)e * TK_;

  // A operand: lane l16 owns rows rowBase + l16 + 16*mi; half selects K sub-pattern.
  const u16* aptr[4];
  #pragma unroll
  for (int mi = 0; mi < 4; ++mi) {
    int ra = rowBase + mi * 16 + l16;
    if (ra > cnt - 1) ra = cnt - 1;
    aptr[mi] = Abf + (size_t)(elist[ra] >> 1) * Kd;
  }

  // B operand: lane owns column nBase + l16 (+ j*16); K-contiguous in memory.
  const u16* bptr = Wbf + ((size_t)e * N + nBase + l16) * Kd + half * 8;

  v8f acc[4][4] = {};

  for (int k0 = 0; k0 < Kd; k0 += 32) {
    V32B A[4];
    #pragma unroll
    for (int mi = 0; mi < 4; ++mi) {
      // lane<16: K = k0+0..7 and k0+16..23 ; lane>=16: K = k0+8..15 and k0+24..31
      A[mi].u[0] = *(const uint4*)(aptr[mi] + k0 + half * 8);
      A[mi].u[1] = *(const uint4*)(aptr[mi] + k0 + 16 + half * 8);
      __builtin_prefetch(aptr[mi] + k0 + 256, 0, 3);   // global_prefetch_b8, A streams
    }
    #pragma unroll
    for (int j = 0; j < 4; ++j) {
      V32B B;
      const uint4* bp = (const uint4*)(bptr + (size_t)j * 16 * Kd + k0);
      B.u[0] = bp[0];
      B.u[1] = bp[2];
      #pragma unroll
      for (int mi = 0; mi < 4; ++mi) {
        acc[mi][j] = __builtin_amdgcn_wmma_f32_16x16x32_bf16(
            /*neg_a=*/false, A[mi].v, /*neg_b=*/false, B.v,
            /*c_mod=*/(short)0, acc[mi][j], /*reuse_a=*/false, /*reuse_b=*/false);
      }
    }
  }

  // D layout: VGPR i -> row i (lanes 0-15) / row i+8 (lanes 16-31), col = l16.
  #pragma unroll
  for (int mi = 0; mi < 4; ++mi) {
    int srow[8];
    #pragma unroll
    for (int i = 0; i < 8; ++i) {
      int r = rowBase + mi * 16 + i + half * 8;
      srow[i] = (r < cnt) ? elist[r] : -1;
    }
    #pragma unroll
    for (int j = 0; j < 4; ++j) {
      int col  = nBase + j * 16 + l16;
      float bj = bias[(size_t)e * N + col];
      #pragma unroll
      for (int i = 0; i < 8; ++i) {
        if (srow[i] >= 0) {
          float v = acc[mi][j][i] + bj;
          if (do_relu) v = fmaxf(v, 0.0f);
          slotY[(size_t)srow[i] * N + col] = v;
        }
      }
    }
  }
}

// ---------------------------------------------------------------------------
// Combine: out[t,n] = rw[t,0]*Y[2t,n] + rw[t,1]*Y[2t+1,n]   (float4 vectorized)
// Stages 1-2 emit packed bf16 activations for the next GEMM; stage 3 emits
// final f32 with ReLU.
// ---------------------------------------------------------------------------
__global__ void k_combine(const float* __restrict__ slotY,
                          const float* __restrict__ rw,     // [T, 2]
                          u16* __restrict__ outB, float* __restrict__ outF,
                          int N, int final_relu) {
  int g = blockIdx.x * blockDim.x + threadIdx.x;   // group of 4 elements
  int nq = N >> 2;
  if (g >= T_ * nq) return;
  int t = g / nq;
  int c = (g - t * nq) << 2;
  float r0 = rw[t * 2 + 0], r1 = rw[t * 2 + 1];
  const float4 y0 = *(const float4*)(slotY + (size_t)(2 * t) * N + c);
  const float4 y1 = *(const float4*)(slotY + (size_t)(2 * t + 1) * N + c);
  float v0 = r0 * y0.x + r1 * y1.x;
  float v1 = r0 * y0.y + r1 * y1.y;
  float v2 = r0 * y0.z + r1 * y1.z;
  float v3 = r0 * y0.w + r1 * y1.w;
  if (final_relu) {
    float4 o;
    o.x = fmaxf(v0, 0.0f); o.y = fmaxf(v1, 0.0f);
    o.z = fmaxf(v2, 0.0f); o.w = fmaxf(v3, 0.0f);
    *(float4*)(outF + (size_t)t * N + c) = o;
  } else {
    uint2 p;
    p.x = (uint32_t)f32_to_bf16(v0) | ((uint32_t)f32_to_bf16(v1) << 16);
    p.y = (uint32_t)f32_to_bf16(v2) | ((uint32_t)f32_to_bf16(v3) << 16);
    *(uint2*)(outB + (size_t)t * N + c) = p;
  }
}

// ---------------------------------------------------------------------------
extern "C" void kernel_launch(void* const* d_in, const int* in_sizes, int n_in,
                              void* d_out, int out_size, void* d_ws, size_t ws_size,
                              hipStream_t stream) {
  (void)in_sizes; (void)n_in; (void)out_size; (void)ws_size;
  const float*         x   = (const float*)d_in[0];
  const unsigned char* m1  = (const unsigned char*)d_in[1];
  const unsigned char* m2  = (const unsigned char*)d_in[2];
  const unsigned char* m3  = (const unsigned char*)d_in[3];
  const float*         rw1 = (const float*)d_in[4];
  const float*         rw2 = (const float*)d_in[5];
  const float*         rw3 = (const float*)d_in[6];
  const float*         W1  = (const float*)d_in[7];
  const float*         b1  = (const float*)d_in[8];
  const float*         W2  = (const float*)d_in[9];
  const float*         b2  = (const float*)d_in[10];
  const float*         W3  = (const float*)d_in[11];
  const float*         b3  = (const float*)d_in[12];
  float* out = (float*)d_out;

  char* ws = (char*)d_ws;
  size_t off = 0;
  auto alloc = [&](size_t bytes) -> void* {
    void* p = ws + off;
    off = (off + bytes + 255) & ~(size_t)255;
    return p;
  };
  u16*   xbf    = (u16*)alloc((size_t)T_ * H_ * 2);
  u16*   h1bf   = (u16*)alloc((size_t)T_ * R0_ * 2);
  u16*   h2bf   = (u16*)alloc((size_t)T_ * R1_ * 2);
  u16*   W1bf   = (u16*)alloc((size_t)E_ * R0_ * H_ * 2);
  u16*   W2bf   = (u16*)alloc((size_t)E_ * R1_ * R0_ * 2);
  u16*   W3bf   = (u16*)alloc((size_t)E_ * O_ * R1_ * 2);
  int*   idx    = (int*)alloc((size_t)E_ * TK_ * 4);
  int*   counts = (int*)alloc((size_t)E_ * 4);
  float* slotY  = (float*)alloc((size_t)TK_ * O_ * 4);

  // One-time conversions to bf16
  {
    int n;
    n = T_ * H_;        k_cvt_bf16<<<(n + 255) / 256, 256, 0, stream>>>(x,  xbf,  n);
    n = E_ * R0_ * H_;  k_cvt_bf16<<<(n + 255) / 256, 256, 0, stream>>>(W1, W1bf, n);
    n = E_ * R1_ * R0_; k_cvt_bf16<<<(n + 255) / 256, 256, 0, stream>>>(W2, W2bf, n);
    n = E_ * O_ * R1_;  k_cvt_bf16<<<(n + 255) / 256, 256, 0, stream>>>(W3, W3bf, n);
  }

  // Stage 1: [T,H] -> [T,R0]
  k_build_lists<<<E_, 256, 0, stream>>>(m1, idx, counts);
  {
    dim3 g(TK_ / 256, R0_ / 64, E_);
    k_moe_gemm<<<g, 128, 0, stream>>>(xbf, W1bf, b1, idx, counts, slotY, H_, R0_, 0);
    int n = T_ * (R0_ / 4);
    k_combine<<<(n + 255) / 256, 256, 0, stream>>>(slotY, rw1, h1bf, nullptr, R0_, 0);
  }

  // Stage 2: [T,R0] -> [T,R1]
  k_build_lists<<<E_, 256, 0, stream>>>(m2, idx, counts);
  {
    dim3 g(TK_ / 256, R1_ / 64, E_);
    k_moe_gemm<<<g, 128, 0, stream>>>(h1bf, W2bf, b2, idx, counts, slotY, R0_, R1_, 0);
    int n = T_ * (R1_ / 4);
    k_combine<<<(n + 255) / 256, 256, 0, stream>>>(slotY, rw2, h2bf, nullptr, R1_, 0);
  }

  // Stage 3: [T,R1] -> [T,O], ReLU on expert output, final ReLU after combine
  k_build_lists<<<E_, 256, 0, stream>>>(m3, idx, counts);
  {
    dim3 g(TK_ / 256, O_ / 64, E_);
    k_moe_gemm<<<g, 128, 0, stream>>>(h2bf, W3bf, b3, idx, counts, slotY, R1_, O_, 1);
    int n = T_ * (O_ / 4);
    k_combine<<<(n + 255) / 256, 256, 0, stream>>>(slotY, rw3, nullptr, out, O_, 1);
  }
}
